// GraphAttentionLayer_45938970198022
// MI455X (gfx1250) — matrix operand order repaired and checked by reading, not compiled, not run on hardware
//
#include <hip/hip_runtime.h>
#include <hip/hip_bf16.h>

#define B_    16
#define N_    2048
#define FIN   256
#define FOUT  128

typedef __attribute__((ext_vector_type(2)))  float  v2f;
typedef __attribute__((ext_vector_type(8)))  float  v8f;
typedef __attribute__((ext_vector_type(16))) __bf16 v16bf;

union V16BF { uint4 q[2]; v16bf v; };
union BF8PK { unsigned short s[8]; uint4 q; };
union BF4PK { __bf16 h[4]; uint2 u; };

// ---------------------------------------------------------------------------
// Phase 1: wh = X @ W^T + b_W  (fp32 WMMA 16x16x4), write whT in bf16
// (layout [B][FOUT][N] so phase-3 B-operand K-runs are contiguous), and
// reduce src = wh.a1, dst = wh.a2 per row (exact, block-local).
// Grid: (N/16, B). Block: 256 = 8 waves; wave w owns h-tile h0 = 16w.
// ---------------------------------------------------------------------------
__global__ __launch_bounds__(256) void gat_phase1(
    const float* __restrict__ X, const float* __restrict__ W,
    const float* __restrict__ bW, const float* __restrict__ a,
    __bf16* __restrict__ whT, float* __restrict__ src, float* __restrict__ dst)
{
  __shared__ float Xs[16 * 260];          // 16 rows x 256 K, stride 260 (bank-safe)
  __shared__ float srcS[16], dstS[16];

  const int t  = threadIdx.x;
  const int b  = blockIdx.y;
  const int i0 = blockIdx.x * 16;

  const float* Xbase = X + ((size_t)b * N_ + i0) * FIN;
  #pragma unroll
  for (int it = 0; it < 4; ++it) {
    int idx4 = t * 4 + it * 1024;                  // 0..4092 step 4
    int row = idx4 >> 8, col = idx4 & 255;
    float4 v = *(const float4*)(Xbase + row * FIN + col);
    *(float4*)(&Xs[row * 260 + col]) = v;
  }
  if (t < 16) { srcS[t] = 0.f; dstS[t] = 0.f; }
  __syncthreads();

  const int lane = t & 31, wave = t >> 5;
  const int h0   = wave * 16;
  const int nIdx = lane & 15;
  const int hiH  = lane >> 4;
  const int koff = hiH * 2;

  const float* Arow = &Xs[nIdx * 260];
  const float* Wcol = W + (size_t)(h0 + nIdx) * FIN;

  v8f acc = {0.f,0.f,0.f,0.f,0.f,0.f,0.f,0.f};
  #pragma unroll 8
  for (int k = 0; k < FIN; k += 4) {
    v2f av = *(const v2f*)(Arow + k + koff);
    v2f bv = *(const v2f*)(Wcol + k + koff);
    acc = __builtin_amdgcn_wmma_f32_16x16x4_f32(
        false, av, false, bv, (short)0, acc, false, false);
  }

  const float bias = bW[h0 + nIdx];
  #pragma unroll
  for (int g = 0; g < 8; ++g) acc[g] += bias;

  {
    BF8PK o;
    #pragma unroll
    for (int g = 0; g < 8; ++g) {
      __bf16 h = (__bf16)acc[g];
      o.s[g] = *(unsigned short*)&h;
    }
    __bf16* p = whT + ((size_t)b * FOUT + h0 + nIdx) * N_ + i0 + hiH * 8;
    *(uint4*)p = o.q;
  }

  const float a1v = a[h0 + nIdx];
  const float a2v = a[FOUT + h0 + nIdx];
  #pragma unroll
  for (int g = 0; g < 8; ++g) {
    float ps = acc[g] * a1v;
    float pd = acc[g] * a2v;
    #pragma unroll
    for (int m = 1; m < 16; m <<= 1) {
      ps += __shfl_xor(ps, m, 16);
      pd += __shfl_xor(pd, m, 16);
    }
    if (nIdx == 0) {
      atomicAdd(&srcS[g + hiH * 8], ps);
      atomicAdd(&dstS[g + hiH * 8], pd);
    }
  }
  __syncthreads();
  if (t < 16) {
    src[(size_t)b * N_ + i0 + t] = srcS[t];
    dst[(size_t)b * N_ + i0 + t] = dstS[t];
  }
}

// ---------------------------------------------------------------------------
// Phase 2: per-batch max of dst (softmax shift upper bound)
// ---------------------------------------------------------------------------
__global__ __launch_bounds__(256) void gat_maxdst(
    const float* __restrict__ dst, float* __restrict__ maxdst)
{
  __shared__ float red[256];
  const int b = blockIdx.x, t = threadIdx.x;
  float m = -3.4e38f;
  for (int j = t; j < N_; j += 256) m = fmaxf(m, dst[(size_t)b * N_ + j]);
  red[t] = m;
  __syncthreads();
  for (int s = 128; s > 0; s >>= 1) {
    if (t < s) red[t] = fmaxf(red[t], red[t + s]);
    __syncthreads();
  }
  if (t == 0) maxdst[b] = red[0];
}

// P stride: 72 bf16 = 144 B = 36 dwords per row; r*36 mod 64 distinct for
// r<16 and divisible by 4 -> conflict-free ds_load_b128 A-operand reads.
#define PSTR 72

// producer: thread handles row r, cols cq..cq+3 of a 16x64 chunk.
__device__ __forceinline__ float produce_chunk(
    const int* __restrict__ adjRow, const float* __restrict__ dstB,
    float srcr, float shiftr, float ba,
    int j0, __bf16* __restrict__ buf, int r, int cq)
{
  const int4   av = *(const int4*)  (adjRow + j0 + cq);
  const float4 dv = *(const float4*)(dstB   + j0 + cq);
  float pv[4];
  const float d[4] = {dv.x, dv.y, dv.z, dv.w};
  const int   m[4] = {av.x, av.y, av.z, av.w};
  float s = 0.f;
  BF4PK pk;
  #pragma unroll
  for (int q = 0; q < 4; ++q) {
    float e = srcr + d[q] + ba;
    e = (e > 0.f) ? e : 0.01f * e;
    pv[q] = (m[q] > 0) ? __expf(e - shiftr) : 0.f;
    s += pv[q];
    pk.h[q] = (__bf16)pv[q];
  }
  *(uint2*)&buf[r * PSTR + cq] = pk.u;   // 8B aligned: r*144 + cq*2, cq%4==0
  return s;
}

// ---------------------------------------------------------------------------
// Phase 3: fused masked-softmax attention + aggregation + ELU.
// 64-j chunks, double-buffered P in LDS, ONE barrier per chunk: next chunk's
// adj/dst global loads are issued before the 2 WMMAs consuming the current
// buffer, so the HBM adj stream overlaps the matrix math.
// Grid: (N/16, B). Block: 256.
// ---------------------------------------------------------------------------
__global__ __launch_bounds__(256) void gat_phase2(
    const int* __restrict__ adj, const __bf16* __restrict__ whT,
    const float* __restrict__ src, const float* __restrict__ dst,
    const float* __restrict__ maxdst, const float* __restrict__ b_a,
    float* __restrict__ out)
{
  __shared__ __bf16 Plds[2][16 * PSTR];
  __shared__ float srcS[16], shiftS[16], lRow[16];

  const int t  = threadIdx.x;
  const int b  = blockIdx.y;
  const int i0 = blockIdx.x * 16;
  const float ba = b_a[0];

  if (t < 16) {
    float s = src[(size_t)b * N_ + i0 + t];
    srcS[t] = s;
    float v = s + maxdst[b] + ba;
    shiftS[t] = (v > 0.f) ? v : 0.01f * v;   // leaky_relu of row-max bound
  }
  __syncthreads();

  // producer indexing: row r, cols cq..cq+3 of each 16x64 chunk
  const int r  = t >> 4;
  const int cq = (t & 15) * 4;
  const float srcr   = srcS[r];
  const float shiftr = shiftS[r];
  const int*   adjRow = adj + ((size_t)b * N_ + i0 + r) * N_;
  const float* dstB   = dst + (size_t)b * N_;

  // consumer indexing (per wave): wave w owns h-cols [16w, 16w+16)
  const int lane = t & 31, wave = t >> 5;
  const int nIdx = lane & 15;
  const int hiH  = lane >> 4;
  const int hcol = wave * 16 + nIdx;
  const __bf16* whRow = whT + ((size_t)b * FOUT + hcol) * N_;
  const int aBase = nIdx * PSTR + hiH * 8;   // + kk*32 (+16) per WMMA

  v8f acc = {0.f,0.f,0.f,0.f,0.f,0.f,0.f,0.f};
  float lpart = 0.f;

  // prologue: fill buffer 0
  lpart += produce_chunk(adjRow, dstB, srcr, shiftr, ba, 0, Plds[0], r, cq);
  __syncthreads();

  int cur = 0;
  for (int j0 = 0; j0 < N_; j0 += 64) {
    const int nxt = cur ^ 1;
    // issue next chunk's global loads before consuming (uniform branch)
    if (j0 + 64 < N_) {
      __builtin_prefetch(adjRow + j0 + 128 + cq, 0, 0);
      lpart += produce_chunk(adjRow, dstB, srcr, shiftr, ba,
                             j0 + 64, Plds[nxt], r, cq);
    }
    // consume current buffer: acc += P(16x64) @ wh(64x16) as 2 bf16 WMMAs
    const __bf16* pb = Plds[cur];
    #pragma unroll
    for (int kk = 0; kk < 2; ++kk) {
      V16BF A, Bv;
      A.q[0] = *(const uint4*)&pb[aBase + kk * 32];
      A.q[1] = *(const uint4*)&pb[aBase + kk * 32 + 16];
      const __bf16* bp = whRow + j0 + kk * 32 + hiH * 8;
      Bv.q[0] = *(const uint4*)bp;
      Bv.q[1] = *(const uint4*)(bp + 16);
      acc = __builtin_amdgcn_wmma_f32_16x16x32_bf16(
          false, A.v, false, Bv.v, (short)0, acc, false, false);
    }
    __syncthreads();
    cur = nxt;
  }

  // row sums: reduce lpart over the 16 producers of each row
  float ls = lpart;
  #pragma unroll
  for (int m = 1; m < 16; m <<= 1) ls += __shfl_xor(ls, m, 16);
  if ((t & 15) == 0) lRow[r] = ls;
  __syncthreads();

  // normalize + ELU + store (C layout: VGPR g, lane-half -> row M)
  #pragma unroll
  for (int g = 0; g < 8; ++g) {
    const int M = g + hiH * 8;
    const float l = lRow[M];
    const float v = acc[g] / l;
    const float o = (v > 0.f) ? v : (__expf(v) - 1.f);
    out[((size_t)b * N_ + i0 + M) * FOUT + hcol] = o;
  }
}

// ---------------------------------------------------------------------------
extern "C" void kernel_launch(void* const* d_in, const int* in_sizes, int n_in,
                              void* d_out, int out_size, void* d_ws, size_t ws_size,
                              hipStream_t stream) {
  const float* X   = (const float*)d_in[0];
  const int*   adj = (const int*)  d_in[1];
  const float* W   = (const float*)d_in[2];
  const float* bW  = (const float*)d_in[3];
  const float* a   = (const float*)d_in[4];
  const float* ba  = (const float*)d_in[5];
  float* out = (float*)d_out;

  char* ws = (char*)d_ws;
  __bf16* whT   = (__bf16*)ws;                              // 16*128*2048*2 = 8 MiB
  float* src    = (float*)(ws + 8388608);                   // 128 KiB
  float* dst    = (float*)(ws + 8388608 + 131072);          // 128 KiB
  float* maxdst = (float*)(ws + 8388608 + 262144);          // 64 B

  dim3 blk(256);
  gat_phase1<<<dim3(N_ / 16, B_), blk, 0, stream>>>(X, W, bW, a, whT, src, dst);
  gat_maxdst<<<B_, blk, 0, stream>>>(dst, maxdst);
  gat_phase2<<<dim3(N_ / 16, B_), blk, 0, stream>>>(adj, whT, src, dst, maxdst, ba, out);
}
